// HyenaOperator_36490042146957
// MI455X (gfx1250) — matrix operand order repaired and checked by reading, not compiled, not run on hardware
//
#include <hip/hip_runtime.h>
#include <hip/hip_bf16.h>
#include <stdint.h>

// ---------------------------------------------------------------------------
// Hyena operator for MI455X (gfx1250, wave32).
//   B=2, L=8192, D=1024, ORDER=2, HID=64
// Pipeline:
//   1) cast x/W_in/W_out/f_w2 -> bf16
//   2) projT = (x @ W_in^T)^T          via WMMA bf16 GEMM (stored transposed)
//   3) per order i:
//        hid  = silu(t*w1+b1)          (tiny, VALU)
//        h_t  = (hid @ f_w2^T + b2)^T  via WMMA GEMM (transposed store)
//        Hf   = rfft(h_t, n=16384)     LDS radix-2 FFT (DIF + untangle)
//        v    = irfft(rfft(v)*Hf)*gate fused FFT-conv kernel, in place in projT
//   4) vT_b = transpose(v) as bf16
//   5) out  = vT_b @ W_out^T           via WMMA GEMM into d_out
// ---------------------------------------------------------------------------

typedef __bf16 bf16_t;
typedef bf16_t v16bf __attribute__((ext_vector_type(16)));
typedef bf16_t v8bf  __attribute__((ext_vector_type(8)));
typedef float  v8f   __attribute__((ext_vector_type(8)));
typedef float  v4f   __attribute__((ext_vector_type(4)));
typedef int    v4i   __attribute__((ext_vector_type(4)));

// address-space qualified v4i for the async load-to-LDS builtin
typedef __attribute__((address_space(1))) v4i as1_v4i;
typedef __attribute__((address_space(3))) v4i as3_v4i;

#if __has_builtin(__builtin_amdgcn_global_load_async_to_lds_b128)
#define HAVE_ASYNC_LDS 1
#endif

__device__ __forceinline__ bf16_t f2bf(float f) {
    // round-to-nearest-even f32 -> bf16
    unsigned u = __float_as_uint(f);
    unsigned r = (u + 0x7FFFu + ((u >> 16) & 1u)) >> 16;
    unsigned short h = (unsigned short)r;
    return __builtin_bit_cast(bf16_t, h);
}

// 16B global -> LDS copy: async (ASYNCcnt, bypasses VGPRs) when available.
__device__ __forceinline__ void stage16(const bf16_t* g, bf16_t* l) {
#ifdef HAVE_ASYNC_LDS
    __builtin_amdgcn_global_load_async_to_lds_b128(
        (as1_v4i*)(uintptr_t)g,
        (as3_v4i*)(uintptr_t)(unsigned)(uintptr_t)l,
        0, 0);
#else
    *(v8bf*)l = *(const v8bf*)g;
#endif
}

__device__ __forceinline__ void stage_wait() {
#ifdef HAVE_ASYNC_LDS
#if __has_builtin(__builtin_amdgcn_s_wait_asynccnt)
    __builtin_amdgcn_s_wait_asynccnt(0);
#else
    asm volatile("s_wait_asynccnt 0x0" ::: "memory");
#endif
#endif
}

// ---------------------------------------------------------------------------
// Generic f32 -> bf16 cast
// ---------------------------------------------------------------------------
__global__ void cast_f32_bf16(const float* __restrict__ in,
                              bf16_t* __restrict__ out, long n) {
    long i = (long)blockIdx.x * blockDim.x + threadIdx.x;
    if (i < n) out[i] = f2bf(in[i]);
}

// ---------------------------------------------------------------------------
// WMMA bf16 GEMM:  C[M,N] = A[M,K] @ B[N,K]^T (+ bias[n])
// A, B row-major bf16 (contraction along contiguous K).
// 128x128 block, 8 waves (256 threads), wave tile 32(M) x 64(N) = 2x4 WMMA
// tiles -> 8 v_wmma per k-step, double-buffered async LDS staging so the
// next k-step's global->LDS copies overlap the current step's WMMAs.
// STORE_T != 0 -> writes C^T[N,M] (8 consecutive m per lane -> b128 stores).
// Requires M%128==0, N%128==0, K%32==0 (holds at all call sites).
// ---------------------------------------------------------------------------
#define TILE_E (128 * 40)   // one 128x32 tile, rows padded 32->40 bf16 (80B)

__device__ __forceinline__ void stage_block(
    const bf16_t* __restrict__ A, const bf16_t* __restrict__ Bm,
    bf16_t* As, bf16_t* Bs, int bm, int bn, int K, int k0, int tid)
{
    #pragma unroll
    for (int p = 0; p < 2; ++p) {
        int idx = p * 256 + tid;           // 0..511
        int row = idx >> 2, ch = idx & 3;  // 4 x 16B chunks per row
        stage16(&A[(size_t)(bm + row) * K + k0 + ch * 8], &As[row * 40 + ch * 8]);
        stage16(&Bm[(size_t)(bn + row) * K + k0 + ch * 8], &Bs[row * 40 + ch * 8]);
    }
}

template <int STORE_T, int HAS_BIAS>
__global__ __launch_bounds__(256) void gemm_bf16_nt(
    const bf16_t* __restrict__ A, const bf16_t* __restrict__ Bm,
    const float* __restrict__ bias, float* __restrict__ C,
    int M, int N, int K)
{
    __shared__ __align__(16) bf16_t AsBuf[2 * TILE_E];
    __shared__ __align__(16) bf16_t BsBuf[2 * TILE_E];

    const int tid  = threadIdx.x;
    const int lane = tid & 31;
    const int wave = tid >> 5;
    const int wm = (wave & 3) * 32;    // wave offset in M within block
    const int wn = (wave >> 2) * 64;   // wave offset in N within block
    const int bm = blockIdx.y * 128;
    const int bn = blockIdx.x * 128;

    const int hi = lane >> 4;          // lane half (0/1)
    const int lr = lane & 15;          // row/col within fragment

    v8f acc[2][4] = {};

    const int nsteps = K >> 5;
    stage_block(A, Bm, &AsBuf[0], &BsBuf[0], bm, bn, K, 0, tid);

    int cur = 0;
    for (int it = 0; it < nsteps; ++it) {
        stage_wait();                  // my async copies into 'cur' are done
        __syncthreads();               // all waves' copies done, last reads done
        if (it + 1 < nsteps)           // prefetch next tile into other buffer
            stage_block(A, Bm, &AsBuf[(cur ^ 1) * TILE_E], &BsBuf[(cur ^ 1) * TILE_E],
                        bm, bn, K, (it + 1) * 32, tid);

        const bf16_t* Asl = &AsBuf[cur * TILE_E];
        const bf16_t* Bsl = &BsBuf[cur * TILE_E];

        // Per-lane fragment gather (ISA 7.12.2 16-bit layouts):
        //   A lane<16: k in {0..7,16..23}; lane>=16: k in {8..15,24..31}
        //   B lane<16: k in {0..7,16..23}; lane>=16: k in {8..15,24..31}
        v16bf afr[2], bfr[4];
        #pragma unroll
        for (int mi = 0; mi < 2; ++mi) {
            int r = wm + mi * 16 + lr;
            v8bf lo  = *(const v8bf*)&Asl[r * 40 + hi * 8];
            v8bf hi8 = *(const v8bf*)&Asl[r * 40 + 16 + hi * 8];
            #pragma unroll
            for (int e = 0; e < 8; ++e) { afr[mi][e] = lo[e]; afr[mi][e + 8] = hi8[e]; }
        }
        #pragma unroll
        for (int ni = 0; ni < 4; ++ni) {
            int r = wn + ni * 16 + lr;
            v8bf lo  = *(const v8bf*)&Bsl[r * 40 + hi * 8];
            v8bf hi8 = *(const v8bf*)&Bsl[r * 40 + 16 + hi * 8];
            #pragma unroll
            for (int e = 0; e < 8; ++e) { bfr[ni][e] = lo[e]; bfr[ni][e + 8] = hi8[e]; }
        }

        #pragma unroll
        for (int mi = 0; mi < 2; ++mi)
            #pragma unroll
            for (int ni = 0; ni < 4; ++ni)
                acc[mi][ni] = __builtin_amdgcn_wmma_f32_16x16x32_bf16(
                    false, afr[mi], false, bfr[ni],
                    (short)0, acc[mi][ni], false, false);
        cur ^= 1;
    }

    // store (C/D layout: lane n = lane&15, VGPR r -> m = r + 8*(lane>=16))
    #pragma unroll
    for (int mi = 0; mi < 2; ++mi) {
        #pragma unroll
        for (int ni = 0; ni < 4; ++ni) {
            int ng = bn + wn + ni * 16 + lr;
            float bv = HAS_BIAS ? bias[ng] : 0.0f;
            if (STORE_T) {
                // this lane's 8 values are m = mgbase..mgbase+7, consecutive
                int mgbase = bm + wm + mi * 16 + hi * 8;
                v4f lo, hi4;
                #pragma unroll
                for (int r = 0; r < 4; ++r)
                    lo[r]  = HAS_BIAS ? acc[mi][ni][r] + bv : acc[mi][ni][r];
                #pragma unroll
                for (int r = 0; r < 4; ++r)
                    hi4[r] = HAS_BIAS ? acc[mi][ni][r + 4] + bv : acc[mi][ni][r + 4];
                *(v4f*)&C[(size_t)ng * M + mgbase]     = lo;
                *(v4f*)&C[(size_t)ng * M + mgbase + 4] = hi4;
            } else {
                #pragma unroll
                for (int r = 0; r < 8; ++r) {
                    int mg = bm + wm + mi * 16 + r + hi * 8;
                    C[(size_t)mg * N + ng] =
                        HAS_BIAS ? acc[mi][ni][r] + bv : acc[mi][ni][r];
                }
            }
        }
    }
}

// ---------------------------------------------------------------------------
// hid[l][u] = silu(t_l * w1[u] + b1[u]), t_l = l/(L-1), stored bf16 (L x 64)
// ---------------------------------------------------------------------------
__global__ void filter_hidden(const float* __restrict__ w1,
                              const float* __restrict__ b1,
                              bf16_t* __restrict__ out, int Ln)
{
    int idx = blockIdx.x * blockDim.x + threadIdx.x;
    if (idx >= Ln * 64) return;
    int l = idx >> 6, u = idx & 63;
    float t = (float)l / (float)(Ln - 1);
    float x = t * w1[u] + b1[u];
    float s = x / (1.0f + __expf(-x));
    out[idx] = f2bf(s);
}

// ---------------------------------------------------------------------------
// FFT helpers: radix-2, N=8192 complex in LDS, real-packed rfft of 16384.
// ---------------------------------------------------------------------------
__device__ __forceinline__ int br13(int x) { return (int)(__brev((unsigned)x) >> 19); }

__device__ __forceinline__ void fft_dif_fwd(float* re, float* im, int tid) {
    // DIF: natural input -> bit-reversed output, e^{-i...} twiddles
    for (int s = 13; s >= 1; --s) {
        int half = 1 << (s - 1);
        for (int t = tid; t < 4096; t += 256) {
            int pos  = t & (half - 1);
            int base = ((t >> (s - 1)) << s) + pos;
            float ar = re[base], ai = im[base];
            float br = re[base + half], bi = im[base + half];
            float theta = -6.2831853071795864f * (float)pos / (float)(1 << s);
            float sn, cs; __sincosf(theta, &sn, &cs);
            re[base] = ar + br;  im[base] = ai + bi;
            float dr = ar - br, di = ai - bi;
            re[base + half] = dr * cs - di * sn;
            im[base + half] = dr * sn + di * cs;
        }
        __syncthreads();
    }
}

__device__ __forceinline__ void fft_dit_inv(float* re, float* im, int tid) {
    // DIT: bit-reversed input -> natural output, e^{+i...} twiddles (no 1/N)
    for (int s = 1; s <= 13; ++s) {
        int half = 1 << (s - 1);
        for (int t = tid; t < 4096; t += 256) {
            int pos  = t & (half - 1);
            int base = ((t >> (s - 1)) << s) + pos;
            float theta = 6.2831853071795864f * (float)pos / (float)(1 << s);
            float sn, cs; __sincosf(theta, &sn, &cs);
            float br = re[base + half], bi = im[base + half];
            float tr = br * cs - bi * sn;
            float ti = br * sn + bi * cs;
            float ar = re[base], ai = im[base];
            re[base] = ar + tr;        im[base] = ai + ti;
            re[base + half] = ar - tr; im[base + half] = ai - ti;
        }
        __syncthreads();
    }
}

// ---------------------------------------------------------------------------
// Hf[d][k] = rfft(h_t[d,:], n=16384), k=0..8192, one workgroup per d.
// ---------------------------------------------------------------------------
__global__ __launch_bounds__(256) void fft_fwd_h(
    const float* __restrict__ hT, float2* __restrict__ Hf, int HFS)
{
    extern __shared__ float lds[];
    float* re = lds;
    float* im = lds + 8192;
    const int tid = threadIdx.x;
    const float* src = hT + (size_t)blockIdx.x * 8192;

    for (int j = tid; j < 8192; j += 256) {        // pack (zero-padded)
        if (j < 4096) { re[j] = src[2 * j]; im[j] = src[2 * j + 1]; }
        else          { re[j] = 0.0f;       im[j] = 0.0f; }
    }
    __syncthreads();
    fft_dif_fwd(re, im, tid);

    float2* out = Hf + (size_t)blockIdx.x * HFS;
    if (tid == 0) {
        float r0 = re[0], i0 = im[0];
        out[0]    = make_float2(r0 + i0, 0.0f);
        out[8192] = make_float2(r0 - i0, 0.0f);
    }
    for (int kk = tid; kk < 4096; kk += 256) {     // untangle pairs (k, N-k)
        int k = kk + 1, m = 8192 - k;
        int pk = br13(k), pm = br13(m);
        float zkr = re[pk], zki = im[pk];
        float zmr = re[pm], zmi = im[pm];
        float Ar = 0.5f * (zkr + zmr), Ai = 0.5f * (zki - zmi);
        float Br = 0.5f * (zkr - zmr), Bi = 0.5f * (zki + zmi);
        float theta = 3.14159265358979f * (float)k / 8192.0f;
        float sn, cs; __sincosf(theta, &sn, &cs);
        float Tr = cs * Br + sn * Bi;
        float Ti = cs * Bi - sn * Br;
        out[k] = make_float2(Ar + Ti,  Ai - Tr);
        out[m] = make_float2(Ar - Ti, -Ai - Tr);
    }
}

// ---------------------------------------------------------------------------
// Fused causal FFT-conv + gate, in place on projT.
// v[b,d,:]  = projT row d,       cols b*8192..b*8192+8191
// gate[b,d] = projT row gc0 + d, same cols
// One workgroup per (b,d): fwd FFT -> spectral *Hf (pairwise at bit-reversed
// positions) -> inv FFT -> *gate -> store.
// ---------------------------------------------------------------------------
__global__ __launch_bounds__(256) void fft_conv_gate(
    float* __restrict__ projT, const float2* __restrict__ Hf,
    int HFS, int gc0)
{
    extern __shared__ float lds[];
    float* re = lds;
    float* im = lds + 8192;
    const int tid = threadIdx.x;
    const int b = blockIdx.x >> 10;
    const int d = blockIdx.x & 1023;
    float*       v = projT + (size_t)d * 16384 + (size_t)b * 8192;
    const float* g = projT + (size_t)(gc0 + d) * 16384 + (size_t)b * 8192;

    for (int j = tid; j < 8192; j += 256) {
        if (j < 4096) { re[j] = v[2 * j]; im[j] = v[2 * j + 1]; }
        else          { re[j] = 0.0f;     im[j] = 0.0f; }
    }
    __syncthreads();
    fft_dif_fwd(re, im, tid);

    // spectral: untangle -> *H -> repack (in place, bit-reversed positions)
    const float2* H = Hf + (size_t)d * HFS;
    if (tid == 0) {
        float r0 = re[0], i0 = im[0];
        float Y0 = (r0 + i0) * H[0].x;
        float YN = (r0 - i0) * H[8192].x;
        re[0] = 0.5f * (Y0 + YN);
        im[0] = 0.5f * (Y0 - YN);
    }
    for (int kk = tid; kk < 4096; kk += 256) {
        int k = kk + 1, m = 8192 - k;
        int pk = br13(k), pm = br13(m);
        float zkr = re[pk], zki = im[pk];
        float zmr = re[pm], zmi = im[pm];
        float Ar = 0.5f * (zkr + zmr), Ai = 0.5f * (zki - zmi);
        float Br = 0.5f * (zkr - zmr), Bi = 0.5f * (zki + zmi);
        float theta = 3.14159265358979f * (float)k / 8192.0f;
        float sn, cs; __sincosf(theta, &sn, &cs);
        float Tr = cs * Br + sn * Bi;
        float Ti = cs * Bi - sn * Br;
        float xkr = Ar + Ti, xki =  Ai - Tr;     // X[k]
        float xmr = Ar - Ti, xmi = -Ai - Tr;     // X[N-k]
        float2 Hk = H[k], Hm = H[m];
        float ykr = xkr * Hk.x - xki * Hk.y;
        float yki = xkr * Hk.y + xki * Hk.x;
        float ymr = xmr * Hm.x - xmi * Hm.y;
        float ymi = xmr * Hm.y + xmi * Hm.x;
        float Pr = 0.5f * (ykr + ymr), Pi = 0.5f * (yki - ymi);
        float Qr = 0.5f * (ykr - ymr), Qi = 0.5f * (yki + ymi);
        float Rr = cs * Qr - sn * Qi;
        float Ri = cs * Qi + sn * Qr;
        re[pk] = Pr - Ri;  im[pk] =  Pi + Rr;    // Z'[k]
        re[pm] = Pr + Ri;  im[pm] = -Pi + Rr;    // Z'[N-k]
    }
    __syncthreads();

    fft_dit_inv(re, im, tid);

    const float invN = 1.0f / 8192.0f;
    for (int j = tid; j < 4096; j += 256) {      // unpack + gate, keep first L
        v[2 * j]     = re[j] * invN * g[2 * j];
        v[2 * j + 1] = im[j] * invN * g[2 * j + 1];
    }
}

// ---------------------------------------------------------------------------
// dst[c][r] = (bf16)src[r][c];  R, C multiples of 32
// ---------------------------------------------------------------------------
__global__ __launch_bounds__(256) void transpose_f32_bf16(
    const float* __restrict__ src, bf16_t* __restrict__ dst, int R, int Cc)
{
    __shared__ float tile[32][33];
    int bx = blockIdx.x * 32;          // col block (in src)
    int by = blockIdx.y * 32;          // row block (in src)
    int tx = threadIdx.x & 31;
    int ty = threadIdx.x >> 5;         // 8 rows per pass
    for (int r = ty; r < 32; r += 8)
        tile[r][tx] = src[(size_t)(by + r) * Cc + bx + tx];
    __syncthreads();
    for (int r = ty; r < 32; r += 8)
        dst[(size_t)(bx + r) * R + by + tx] = f2bf(tile[tx][r]);
}

// ---------------------------------------------------------------------------
// Host-side orchestration
// ---------------------------------------------------------------------------
extern "C" void kernel_launch(void* const* d_in, const int* in_sizes, int n_in,
                              void* d_out, int out_size, void* d_ws, size_t ws_size,
                              hipStream_t stream) {
    (void)in_sizes; (void)n_in; (void)out_size; (void)ws_size;
    constexpr int Bn = 2, L = 8192, Dn = 1024, ORDER = 2, HID = 64;
    constexpr int Mtot = Bn * L;            // 16384
    constexpr int N3 = (ORDER + 1) * Dn;    // 3072
    constexpr int HFS = 8200;               // Hf row stride (float2 units)

    const float* x     = (const float*)d_in[0];
    const float* W_in  = (const float*)d_in[1];
    const float* W_out = (const float*)d_in[2];
    const float* f_w1  = (const float*)d_in[3];
    const float* f_b1  = (const float*)d_in[4];
    const float* f_w2  = (const float*)d_in[5];
    const float* f_b2  = (const float*)d_in[6];
    float* out = (float*)d_out;

    // workspace carve
    char* w = (char*)d_ws;
    size_t off = 0;
    auto carve = [&](size_t bytes) -> void* {
        void* p = w + off;
        off = (off + bytes + 255) & ~(size_t)255;
        return p;
    };
    bf16_t* xb     = (bf16_t*)carve((size_t)Mtot * Dn * 2);
    bf16_t* Winb   = (bf16_t*)carve((size_t)N3 * Dn * 2);
    bf16_t* Woutb  = (bf16_t*)carve((size_t)Dn * Dn * 2);
    bf16_t* fw2b   = (bf16_t*)carve((size_t)ORDER * Dn * HID * 2);
    bf16_t* hidb   = (bf16_t*)carve((size_t)L * HID * 2);
    float*  h_t    = (float*)carve((size_t)Dn * L * 4);
    float2* Hf     = (float2*)carve((size_t)Dn * HFS * 8);
    float*  projT  = (float*)carve((size_t)N3 * Mtot * 4);
    bf16_t* vTb    = (bf16_t*)carve((size_t)Mtot * Dn * 2);

    auto cast = [&](const float* src, bf16_t* dst, long n) {
        cast_f32_bf16<<<dim3((unsigned)((n + 255) / 256)), dim3(256), 0, stream>>>(src, dst, n);
    };
    cast(x,     xb,    (long)Mtot * Dn);
    cast(W_in,  Winb,  (long)N3 * Dn);
    cast(W_out, Woutb, (long)Dn * Dn);
    cast(f_w2,  fw2b,  (long)ORDER * Dn * HID);

    // projT = (x @ W_in^T)^T : rows = channel splits, cols = (b,l) -- unit
    // stride sequences for all downstream FFT work.
    gemm_bf16_nt<1, 0><<<dim3(N3 / 128, Mtot / 128), dim3(256), 0, stream>>>(
        xb, Winb, nullptr, projT, Mtot, N3, Dn);

    for (int i = 0; i < ORDER; ++i) {
        filter_hidden<<<dim3(L * HID / 256), dim3(256), 0, stream>>>(
            f_w1 + i * HID, f_b1 + i * HID, hidb, L);
        // h_t[d][l] = (hid @ f_w2[i]^T + f_b2[i])^T
        gemm_bf16_nt<1, 1><<<dim3(Dn / 128, L / 128), dim3(256), 0, stream>>>(
            hidb, fw2b + (size_t)i * Dn * HID, f_b2 + i * Dn, h_t,
            L, Dn, HID);
        fft_fwd_h<<<dim3(Dn), dim3(256), 65536, stream>>>(h_t, Hf, HFS);
        fft_conv_gate<<<dim3(Bn * Dn), dim3(256), 65536, stream>>>(
            projT, Hf, HFS, (i + 1) * Dn);
    }

    // vTb[(b,l)][d] = projT[d][(b,l)]  (bf16)
    transpose_f32_bf16<<<dim3(Mtot / 32, Dn / 32), dim3(256), 0, stream>>>(
        projT, vTb, Dn, Mtot);

    // out = v @ W_out^T
    gemm_bf16_nt<0, 0><<<dim3(Dn / 128, Mtot / 128), dim3(256), 0, stream>>>(
        vTb, Woutb, nullptr, out, Mtot, Dn, Dn);
}
